// Net_74079595921835
// MI455X (gfx1250) — matrix operand order repaired and checked by reading, not compiled, not run on hardware
//
#include <hip/hip_runtime.h>
#include <hip/hip_bf16.h>
#include <math.h>

// ---------------- problem constants (match reference) ----------------
#define BATCH   256
#define N0      1000
#define EDG     16000
#define DD      32
#define K1      800
#define K2      640
#define K3      512

typedef __attribute__((ext_vector_type(16))) _Float16 v16h;
typedef __attribute__((ext_vector_type(8)))  float    v8f;

#define CDIV(a,b) (((a)+(b)-1)/(b))

// =====================================================================
// WMMA fragment loaders (wave32, CDNA5 ISA 7.12.2 layouts).
// All row counts are exact multiples of 16, so no bounds guards needed:
// straight-line loads -> global_load_b128 + v_cvt_pk_f16_f32 packing.
// =====================================================================

// 16-bit A matrix 16x32 (MxK), f32 source converted to f16.
// lane 0-15: M=lane; VGPR v: v<4 -> K=2v..2v+1 ; v>=4 -> K=16+2(v-4)..
// lane16-31: same M, K offset +8 within each group of 4 VGPRs.
__device__ __forceinline__ v16h load_afrag(const float* __restrict__ X,
                                           int m0, int lane, int ldx) {
  v16h a;
  int M  = m0 + (lane & 15);
  int hi = lane >> 4;
  const float* xr = X + (size_t)M * ldx;
#pragma unroll
  for (int v = 0; v < 8; ++v) {
    int K = ((v >> 2) << 4) + (hi << 3) + ((v & 3) << 1);
    a[2 * v]     = (_Float16)xr[K];
    a[2 * v + 1] = (_Float16)xr[K + 1];
  }
  return a;
}

// 16-bit B matrix 32x16 (KxN) where B[k][n] = Wrow(j0+n)[koff+k],
// Wrow has leading dimension ldw. K layout same as A fragment (table order).
__device__ __forceinline__ v16h load_bfrag(const float* __restrict__ W,
                                           int j0, int koff, int lane, int ldw) {
  v16h b;
  int N  = lane & 15;
  int hi = lane >> 4;
  const float* wr = W + (size_t)(j0 + N) * ldw + koff;
#pragma unroll
  for (int v = 0; v < 8; ++v) {
    int K = ((v >> 2) << 4) + (hi << 3) + ((v & 3) << 1);
    b[2 * v]     = (_Float16)wr[K];
    b[2 * v + 1] = (_Float16)wr[K + 1];
  }
  return b;
}

// =====================================================================
// Kernel: h = relu(x @ W^T + b); also aggr = h (self-loop seed for max)
// X: [rows,32] contiguous, W: [32,32], bias: [32].
// One wave => one 16x16 output tile; two N-tiles cover the 32 outputs.
// Grid sized exactly (rows % 64 == 0) => no tail, no guards.
// =====================================================================
__global__ void lin_relu_wmma(const float* __restrict__ X,
                              const float* __restrict__ W,
                              const float* __restrict__ bias,
                              float* __restrict__ H,
                              float* __restrict__ AG) {
  int gwave = (blockIdx.x * blockDim.x + threadIdx.x) >> 5;
  int lane  = threadIdx.x & 31;
  int m0 = (gwave >> 1) << 4;
  int j0 = (gwave & 1) << 4;

  v16h a  = load_afrag(X, m0, lane, DD);
  v16h bb = load_bfrag(W, j0, 0, lane, DD);
  float bv = bias[j0 + (lane & 15)];
  v8f c;
#pragma unroll
  for (int r = 0; r < 8; ++r) c[r] = bv;

  v8f d = __builtin_amdgcn_wmma_f32_16x16x32_f16(false, a, false, bb,
                                                 (short)0, c, false, false);
  int N = lane & 15, hi = lane >> 4;
#pragma unroll
  for (int r = 0; r < 8; ++r) {
    int M = m0 + (hi << 3) + r;
    float v = fmaxf(d[r], 0.f);
    size_t off = (size_t)M * DD + j0 + N;
    H[off]  = v;
    AG[off] = v;                                   // self-loop seed
  }
}

// =====================================================================
// Kernel: edge max-aggregation. aggr[dst] = max(aggr[dst], h[src]).
// All values >= 0 (post-ReLU) so uint bit-pattern atomicMax == float max.
// 4 threads per edge, 8 features each. L2 (192 MB) holds the whole
// 32 MB h-buffer; prefetch hides the irregular gather.
// =====================================================================
__global__ void edge_max_aggr(const int* __restrict__ edges,
                              const float* __restrict__ H,
                              float* __restrict__ AG,
                              int n) {
  long long t = (long long)blockIdx.x * blockDim.x + threadIdx.x;
  long long total = (long long)BATCH * EDG * 4;
  if (t >= total) return;
  int fq = (int)(t & 3);
  long long eg = t >> 2;
  int b = (int)(eg / EDG);
  int e = (int)(eg % EDG);
  const int* eb = edges + (size_t)b * 2 * EDG;
  int s = eb[e];
  if (s < 0) return;                               // invalid edge
  int dd = eb[EDG + e];
  const float* hs = H + ((size_t)b * n + s) * DD + fq * 8;
  __builtin_prefetch(hs, 0, 3);                    // global_prefetch into L2/L0
  unsigned int* ad = (unsigned int*)(AG + ((size_t)b * n + dd) * DD + fq * 8);
#pragma unroll
  for (int f = 0; f < 8; ++f) {
    atomicMax(&ad[f], __float_as_uint(hs[f]));
  }
}

// =====================================================================
// Kernel: xo = relu(concat([aggr, x]) @ U^T)   (U: [32,64], no bias)
// Two accumulating WMMAs: K=0..31 from aggr, K=32..63 from x.
// =====================================================================
__global__ void sage_update_wmma(const float* __restrict__ AG,
                                 const float* __restrict__ X,
                                 const float* __restrict__ U,
                                 float* __restrict__ XO) {
  int gwave = (blockIdx.x * blockDim.x + threadIdx.x) >> 5;
  int lane  = threadIdx.x & 31;
  int m0 = (gwave >> 1) << 4;
  int j0 = (gwave & 1) << 4;

  v16h a0 = load_afrag(AG, m0, lane, DD);
  v16h a1 = load_afrag(X,  m0, lane, DD);
  v16h b0 = load_bfrag(U, j0, 0,  lane, 2 * DD);
  v16h b1 = load_bfrag(U, j0, DD, lane, 2 * DD);
  v8f c;
#pragma unroll
  for (int r = 0; r < 8; ++r) c[r] = 0.f;

  c = __builtin_amdgcn_wmma_f32_16x16x32_f16(false, a0, false, b0,
                                             (short)0, c, false, false);
  c = __builtin_amdgcn_wmma_f32_16x16x32_f16(false, a1, false, b1,
                                             (short)0, c, false, false);
  int N = lane & 15, hi = lane >> 4;
#pragma unroll
  for (int r = 0; r < 8; ++r) {
    int M = m0 + (hi << 3) + r;
    XO[(size_t)M * DD + j0 + N] = fmaxf(c[r], 0.f);
  }
}

// =====================================================================
// Kernel: TopK pooling + readout accumulation. One block (256 thr) per graph.
// scores -> LDS bitonic sort (descending, 1024 slots) -> top-k gather,
// newidx scatter, gmp||gap accumulated into Eacc[b][0:64].
// =====================================================================
__global__ void topk_pool(const float* __restrict__ X,   // [B, n, 32]
                          float* __restrict__ XP,        // [B, k, 32]
                          const float* __restrict__ p,   // [32]
                          int n, int k,
                          int* __restrict__ newidx,      // [B, 1024]
                          float* __restrict__ Eacc) {    // [B, 64]
  __shared__ float sc[1024];
  __shared__ int   si[1024];
  __shared__ float pv[DD];
  __shared__ float rmax[8][DD];
  __shared__ float rsum[8][DD];

  int b = blockIdx.x, tid = threadIdx.x;
  if (tid < DD) pv[tid] = p[tid];
  __syncthreads();

  float nrm = 0.f;
#pragma unroll
  for (int i = 0; i < DD; ++i) nrm += pv[i] * pv[i];
  nrm = sqrtf(nrm);

  const float* xb = X + (size_t)b * n * DD;
  for (int i = tid; i < 1024; i += 256) {
    float s;
    if (i < n) {
      const float* xr = xb + (size_t)i * DD;
      float dot = 0.f;
#pragma unroll
      for (int f = 0; f < DD; ++f) dot += xr[f] * pv[f];
      s = tanhf(dot / nrm);
    } else {
      s = -1e30f;
    }
    sc[i] = s; si[i] = i;
  }
  __syncthreads();

  // bitonic sort, 1024 elements, descending overall (j is a power of two:
  // pair index via masks, no integer division)
  for (int kk = 2; kk <= 1024; kk <<= 1) {
    for (int j = kk >> 1; j > 0; j >>= 1) {
      int jm = j - 1;
      for (int t = tid; t < 512; t += 256) {
        int i   = ((t & ~jm) << 1) | (t & jm);
        int ixj = i + j;
        bool desc = ((i & kk) == 0);
        float a0 = sc[i], a1 = sc[ixj];
        bool sw = desc ? (a0 < a1) : (a0 > a1);
        if (sw) {
          sc[i] = a1; sc[ixj] = a0;
          int t0 = si[i]; si[i] = si[ixj]; si[ixj] = t0;
        }
      }
      __syncthreads();
    }
  }

  // newidx: old -> new (or -1)
  int* ni = newidx + (size_t)b * 1024;
  for (int i = tid; i < n; i += 256) ni[i] = -1;
  __syncthreads();
  for (int i = tid; i < k; i += 256) ni[si[i]] = i;

  // pooled features xp[i] = x[perm[i]] * vals[i]
  float* xpb = XP + (size_t)b * k * DD;
  for (int i = tid; i < k; i += 256) {
    float v = sc[i];
    const float* xr = xb + (size_t)si[i] * DD;
    float* xo = xpb + (size_t)i * DD;
#pragma unroll
    for (int f = 0; f < DD; ++f) xo[f] = xr[f] * v;
  }
  __syncthreads();

  // readout gmp || gap over pooled nodes, accumulate into Eacc
  int f = tid & 31, c = tid >> 5;
  float mx = -1e30f, sm = 0.f;
  for (int i = c; i < k; i += 8) {
    float v = xpb[(size_t)i * DD + f];
    mx = fmaxf(mx, v); sm += v;
  }
  rmax[c][f] = mx; rsum[c][f] = sm;
  __syncthreads();
  if (tid < DD) {
    float m = rmax[0][tid], s = rsum[0][tid];
#pragma unroll
    for (int cc = 1; cc < 8; ++cc) { m = fmaxf(m, rmax[cc][tid]); s += rsum[cc][tid]; }
    Eacc[(size_t)b * 64 + tid]      += m;
    Eacc[(size_t)b * 64 + 32 + tid] += s / (float)k;
  }
}

// =====================================================================
// Kernel: reindex edges through newidx; invalid edges marked src = -1.
// =====================================================================
__global__ void edge_remap(int* __restrict__ edges,
                           const int* __restrict__ newidx) {
  long long t = (long long)blockIdx.x * blockDim.x + threadIdx.x;
  if (t >= (long long)BATCH * EDG) return;
  int b = (int)(t / EDG);
  int e = (int)(t % EDG);
  int* eb = edges + (size_t)b * 2 * EDG;
  int s = eb[e];
  if (s < 0) return;
  int d = eb[EDG + e];
  const int* ni = newidx + (size_t)b * 1024;
  int s2 = ni[s], d2 = ni[d];
  if (s2 < 0 || d2 < 0) { eb[e] = -1; eb[EDG + e] = -1; }
  else                  { eb[e] = s2; eb[EDG + e] = d2; }
}

// =====================================================================
// Kernel: embedding gather x = emb[ids]
// =====================================================================
__global__ void embed_gather(const int* __restrict__ ids,
                             const float* __restrict__ emb,
                             float* __restrict__ X) {
  long long t = (long long)blockIdx.x * blockDim.x + threadIdx.x;
  if (t >= (long long)BATCH * N0) return;
  int id = ids[t];
  const float* er = emb + (size_t)id * DD;
  float* xr = X + (size_t)t * DD;
#pragma unroll
  for (int f = 0; f < DD; ++f) xr[f] = er[f];
}

// =====================================================================
// Kernel: per-launch init (deterministic): copy edges, zero Eacc.
// =====================================================================
__global__ void init_ws(const int* __restrict__ ei,
                        int* __restrict__ edges,
                        float* __restrict__ Eacc) {
  long long t = (long long)blockIdx.x * blockDim.x + threadIdx.x;
  long long ecnt = (long long)BATCH * 2 * EDG;
  if (t < ecnt) edges[t] = ei[t];
  if (t < (long long)BATCH * 64) Eacc[t] = 0.f;
}

// =====================================================================
// Kernel: MLP head. One wave per graph. out = [o(256) | embed(256*64)]
// =====================================================================
__global__ void mlp_head(const float* __restrict__ Eacc,
                         const float* __restrict__ l1w, const float* __restrict__ l1b,
                         const float* __restrict__ l2w, const float* __restrict__ l2b,
                         const float* __restrict__ l3w, const float* __restrict__ l3b,
                         float* __restrict__ out) {
  int b = blockIdx.x, lane = threadIdx.x;   // 32 threads
  __shared__ float ev[64], h1[32], h2[16];
  ev[lane]      = Eacc[(size_t)b * 64 + lane];
  ev[lane + 32] = Eacc[(size_t)b * 64 + 32 + lane];
  __syncthreads();

  float acc = l1b[lane];
#pragma unroll 8
  for (int kk = 0; kk < 64; ++kk) acc += ev[kk] * l1w[(size_t)lane * 64 + kk];
  h1[lane] = fmaxf(acc, 0.f);
  __syncthreads();

  if (lane < 16) {
    float a2 = l2b[lane];
#pragma unroll 8
    for (int kk = 0; kk < 32; ++kk) a2 += h1[kk] * l2w[(size_t)lane * 32 + kk];
    h2[lane] = fmaxf(a2, 0.f);
  }
  __syncthreads();

  if (lane == 0) {
    float a3 = l3b[0];
#pragma unroll
    for (int kk = 0; kk < 16; ++kk) a3 += h2[kk] * l3w[kk];
    out[b] = 1.f / (1.f + expf(-a3));
  }
  out[BATCH + (size_t)b * 64 + lane]      = ev[lane];
  out[BATCH + (size_t)b * 64 + 32 + lane] = ev[lane + 32];
}

// =====================================================================
// Host launch
// =====================================================================
extern "C" void kernel_launch(void* const* d_in, const int* in_sizes, int n_in,
                              void* d_out, int out_size, void* d_ws, size_t ws_size,
                              hipStream_t stream) {
  const int*   x_ids = (const int*)  d_in[0];
  const int*   ei    = (const int*)  d_in[1];
  const float* emb   = (const float*)d_in[2];
  const float* Wl[3] = {(const float*)d_in[3],  (const float*)d_in[7],  (const float*)d_in[11]};
  const float* Bl[3] = {(const float*)d_in[4],  (const float*)d_in[8],  (const float*)d_in[12]};
  const float* Ul[3] = {(const float*)d_in[5],  (const float*)d_in[9],  (const float*)d_in[13]};
  const float* Pl[3] = {(const float*)d_in[6],  (const float*)d_in[10], (const float*)d_in[14]};
  const float* l1w = (const float*)d_in[15];
  const float* l1b = (const float*)d_in[16];
  const float* l2w = (const float*)d_in[17];
  const float* l2b = (const float*)d_in[18];
  const float* l3w = (const float*)d_in[19];
  const float* l3b = (const float*)d_in[20];
  float* out = (float*)d_out;

  // ---- workspace carve-up ----
  char* ws = (char*)d_ws;
  const size_t NODEBUF = (size_t)BATCH * N0 * DD * sizeof(float);   // 32.77 MB
  float* xcur  = (float*)ws;                 ws += NODEBUF;
  float* xalt  = (float*)ws;                 ws += NODEBUF;
  float* hbuf  = (float*)ws;                 ws += NODEBUF;
  float* aggr  = (float*)ws;                 ws += NODEBUF;
  int*   edges = (int*)ws;                   ws += (size_t)BATCH * 2 * EDG * sizeof(int);
  int*   nidx  = (int*)ws;                   ws += (size_t)BATCH * 1024 * sizeof(int);
  float* Eacc  = (float*)ws;                 ws += (size_t)BATCH * 64 * sizeof(float);

  // ---- init (copy edges, zero Eacc) + embedding gather ----
  {
    long long ecnt = (long long)BATCH * 2 * EDG;
    init_ws<<<(unsigned)CDIV(ecnt, 256), 256, 0, stream>>>(ei, edges, Eacc);
    long long ncnt = (long long)BATCH * N0;
    embed_gather<<<(unsigned)CDIV(ncnt, 256), 256, 0, stream>>>(x_ids, emb, xcur);
  }

  const int ns[3] = {N0, K1, K2};
  const int ks[3] = {K1, K2, K3};

  for (int l = 0; l < 3; ++l) {
    int n = ns[l], k = ks[l];
    int rows = BATCH * n;
    // rows % 64 == 0 for all layers: exact grid, guard-free GEMM kernels.
    unsigned gemmBlocks = (unsigned)(rows / 64);   // (rows/16 tiles)*2 waves / 8 waves per block

    // h = relu(x@W^T+b), aggr = h (self-loop)
    lin_relu_wmma<<<gemmBlocks, 256, 0, stream>>>(xcur, Wl[l], Bl[l], hbuf, aggr);

    // scatter-max over edges
    long long at = (long long)BATCH * EDG * 4;
    edge_max_aggr<<<(unsigned)CDIV(at, 256), 256, 0, stream>>>(edges, hbuf, aggr, n);

    // x' = relu(concat([aggr, x]) @ U^T)
    sage_update_wmma<<<gemmBlocks, 256, 0, stream>>>(aggr, xcur, Ul[l], xalt);

    // top-k pool: reads xalt (stride n), writes xcur (stride k), readout into Eacc
    topk_pool<<<BATCH, 256, 0, stream>>>(xalt, xcur, Pl[l], n, k, nidx, Eacc);

    // reindex edges through newidx
    long long et = (long long)BATCH * EDG;
    edge_remap<<<(unsigned)CDIV(et, 256), 256, 0, stream>>>(edges, nidx);
  }

  // ---- MLP head + embedding output ----
  mlp_head<<<BATCH, 32, 0, stream>>>(Eacc, l1w, l1b, l2w, l2b, l3w, l3b, out);

  (void)in_sizes; (void)n_in; (void)out_size; (void)ws_size;
}
